// LSTMSemRel_Prod_31705448579592
// MI455X (gfx1250) — compile-verified, matrix operand order checked by reading, not run on hardware
//
#include <hip/hip_runtime.h>
#include <hip/hip_bf16.h>

typedef __attribute__((ext_vector_type(16))) _Float16 v16h;
typedef __attribute__((ext_vector_type(8)))  _Float16 v8h;
typedef __attribute__((ext_vector_type(4)))  _Float16 v4h;
typedef __attribute__((ext_vector_type(8)))  float    v8f;

#define LDH 40   // LDS row stride in halfs (80B -> conflict-free b128 frag loads)
#define BM 128
#define BN 64
#define BK 32

__device__ __forceinline__ float4 ld4(const float* p) { return *(const float4*)p; }

// ---------------------------------------------------------------------------
// Tiled WMMA GEMM for one LSTM step:  G[M x 1024] = [X_t | H] @ [Wi | Wh]^T
// MODE 0 (word):  X_t = embed_table[tokens[:, t]]  (E=300), H = h_w (256), K=556 (pad 576)
// MODE 1 (sent):  X_t = sent_h[:, s, :]            (256),   H = h_s (256), K=512
// Block tile 128x64, 8 waves (4M x 2N), each wave -> 32x32 via 4x v_wmma_f32_16x16x32_f16
// K loop split into uniform phases so the hot chunks have no boundary cndmasks,
// use global b128 loads (aligned phases) and packed b64 LDS stores.
// ---------------------------------------------------------------------------
template<int MODE>
__global__ __launch_bounds__(256) void lstm_gemm_kernel(
    const int*   __restrict__ tokens,
    const float* __restrict__ xsrc,    // embed table (MODE0) or sent_h (MODE1)
    const float* __restrict__ hbuf,
    const float* __restrict__ Wi,
    const float* __restrict__ Wh,
    float*       __restrict__ G,
    int t)
{
    __shared__ __align__(16) _Float16 As[BM * LDH];
    __shared__ __align__(16) _Float16 Bs[BN * LDH];

    const int tid  = threadIdx.x;
    const int lane = tid & 31;
    const int wave = tid >> 5;
    const int wm = wave & 3;          // 4 waves along M
    const int wn = wave >> 2;         // 2 waves along N
    const int Mbase = blockIdx.y * BM;
    const int Nbase = blockIdx.x * BN;

    v8f acc[2][2] = {};

    const int cg = (tid & 7) * 4;     // 4-col group within the K chunk
    const int rg = tid >> 3;          // row group 0..31
    const int g  = lane >> 4;         // K-half select for fragments
    const int l  = lane & 15;         // row/col within 16x16 tile

    // One K-chunk: stage tiles via caller-provided element loaders, then 4 WMMAs.
    auto do_chunk = [&](auto loadA4, auto loadB4) {
        #pragma unroll
        for (int i = 0; i < 4; ++i) {
            const int row = rg + i * 32;
            const float4 v = loadA4(Mbase + row);
            v4h p; p[0] = (_Float16)v.x; p[1] = (_Float16)v.y;
                   p[2] = (_Float16)v.z; p[3] = (_Float16)v.w;
            *(v4h*)&As[row * LDH + cg] = p;
        }
        #pragma unroll
        for (int i = 0; i < 2; ++i) {
            const int row = rg + i * 32;
            const float4 v = loadB4(Nbase + row);
            v4h p; p[0] = (_Float16)v.x; p[1] = (_Float16)v.y;
                   p[2] = (_Float16)v.z; p[3] = (_Float16)v.w;
            *(v4h*)&Bs[row * LDH + cg] = p;
        }
        __syncthreads();

        v16h afrag[2], bfrag[2];
        #pragma unroll
        for (int sm = 0; sm < 2; ++sm) {
            const int row = wm * 32 + sm * 16 + l;
            const v8h lo = *(const v8h*)&As[row * LDH + 8 * g];        // K = 8g..8g+7
            const v8h hi = *(const v8h*)&As[row * LDH + 16 + 8 * g];   // K = 16+8g..
            #pragma unroll
            for (int e = 0; e < 8; ++e) { afrag[sm][e] = lo[e]; afrag[sm][e + 8] = hi[e]; }
        }
        #pragma unroll
        for (int sn = 0; sn < 2; ++sn) {
            const int row = wn * 32 + sn * 16 + l;
            const v8h lo = *(const v8h*)&Bs[row * LDH + 8 * g];
            const v8h hi = *(const v8h*)&Bs[row * LDH + 16 + 8 * g];
            #pragma unroll
            for (int e = 0; e < 8; ++e) { bfrag[sn][e] = lo[e]; bfrag[sn][e + 8] = hi[e]; }
        }
        #pragma unroll
        for (int sm = 0; sm < 2; ++sm)
            #pragma unroll
            for (int sn = 0; sn < 2; ++sn)
                acc[sm][sn] = __builtin_amdgcn_wmma_f32_16x16x32_f16(
                    false, afrag[sm], false, bfrag[sn],
                    (short)0, acc[sm][sn], false, false);
        __syncthreads();
    };

    if constexpr (MODE == 0) {
        // ---- phase E: chunks 0..8, pure embedding (k in [0,288)) ----
        for (int kc = 0; kc < 9; ++kc) {
            const int kcol = kc * BK + cg;
            do_chunk(
                [&](int m) {
                    const int tok = tokens[m * 32 + t];
                    const float* p = &xsrc[tok * 300 + kcol];
                    return make_float4(p[0], p[1], p[2], p[3]);
                },
                [&](int n) {
                    const float* p = &Wi[n * 300 + kcol];
                    return make_float4(p[0], p[1], p[2], p[3]);
                });
        }
        // ---- mixed chunk 9: k in [288,320), boundary at 300 ----
        {
            const int kcol = 288 + cg;
            do_chunk(
                [&](int m) {
                    const int tok = tokens[m * 32 + t];
                    float r[4];
                    #pragma unroll
                    for (int c = 0; c < 4; ++c) {
                        const int k = kcol + c;
                        r[c] = (k < 300) ? xsrc[tok * 300 + k] : hbuf[m * 256 + (k - 300)];
                    }
                    return make_float4(r[0], r[1], r[2], r[3]);
                },
                [&](int n) {
                    float r[4];
                    #pragma unroll
                    for (int c = 0; c < 4; ++c) {
                        const int k = kcol + c;
                        r[c] = (k < 300) ? Wi[n * 300 + k] : Wh[n * 256 + (k - 300)];
                    }
                    return make_float4(r[0], r[1], r[2], r[3]);
                });
        }
        // ---- phase H: chunks 10..16, pure hidden (k-300 in [20,244), 16B aligned) ----
        for (int kc = 10; kc < 17; ++kc) {
            const int koff = kc * BK - 300 + cg;
            do_chunk(
                [&](int m) { return ld4(&hbuf[m * 256 + koff]); },
                [&](int n) {
                    if (koff + 32 < 256) __builtin_prefetch(&Wh[n * 256 + koff + 32], 0, 1);
                    return ld4(&Wh[n * 256 + koff]);
                });
        }
        // ---- tail chunk 17: k in [544,576), valid h up to k<556, zero pad after ----
        {
            const int koff = 244 + cg;   // k - 300
            do_chunk(
                [&](int m) {
                    float r[4];
                    #pragma unroll
                    for (int c = 0; c < 4; ++c)
                        r[c] = (koff + c < 256) ? hbuf[m * 256 + koff + c] : 0.f;
                    return make_float4(r[0], r[1], r[2], r[3]);
                },
                [&](int n) {
                    float r[4];
                    #pragma unroll
                    for (int c = 0; c < 4; ++c)
                        r[c] = (koff + c < 256) ? Wh[n * 256 + koff + c] : 0.f;
                    return make_float4(r[0], r[1], r[2], r[3]);
                });
        }
    } else {
        // ---- phase X: chunks 0..7, x-half (all 16B aligned) ----
        for (int kc = 0; kc < 8; ++kc) {
            const int kcol = kc * BK + cg;
            do_chunk(
                [&](int m) { return ld4(&xsrc[(m * 8 + t) * 256 + kcol]); },
                [&](int n) {
                    if (kcol + 32 < 256) __builtin_prefetch(&Wi[n * 256 + kcol + 32], 0, 1);
                    return ld4(&Wi[n * 256 + kcol]);
                });
        }
        // ---- phase H: chunks 8..15, h-half ----
        for (int kc = 8; kc < 16; ++kc) {
            const int koff = kc * BK - 256 + cg;
            do_chunk(
                [&](int m) { return ld4(&hbuf[m * 256 + koff]); },
                [&](int n) {
                    if (koff + 32 < 256) __builtin_prefetch(&Wh[n * 256 + koff + 32], 0, 1);
                    return ld4(&Wh[n * 256 + koff]);
                });
        }
    }

    // ---- epilogue: C/D layout -> G (raw gates; bias added in gate kernel) ----
    #pragma unroll
    for (int sm = 0; sm < 2; ++sm)
        #pragma unroll
        for (int sn = 0; sn < 2; ++sn)
            #pragma unroll
            for (int r = 0; r < 8; ++r) {
                const int row = Mbase + wm * 32 + sm * 16 + r + 8 * g;
                const int col = Nbase + wn * 32 + sn * 16 + l;
                G[row * 1024 + col] = acc[sm][sn][r];
            }
}

// ---------------------------------------------------------------------------
// Gate nonlinearity + state update.  g split order: i, f, g, o.
// ---------------------------------------------------------------------------
__global__ __launch_bounds__(256) void lstm_gate_kernel(
    const float* __restrict__ G,
    const float* __restrict__ bi,
    const float* __restrict__ bh,
    const int*   __restrict__ lengths,
    float* __restrict__ h,
    float* __restrict__ c,
    int t, int useMask)
{
    const int idx = blockIdx.x * 256 + threadIdx.x;
    const int n = idx >> 8;
    const int d = idx & 255;
    const float* g = G + n * 1024;
    const float gi = g[d]       + bi[d]       + bh[d];
    const float gf = g[256 + d] + bi[256 + d] + bh[256 + d];
    const float gc = g[512 + d] + bi[512 + d] + bh[512 + d];
    const float go = g[768 + d] + bi[768 + d] + bh[768 + d];
    const float i_ = 1.f / (1.f + __expf(-gi));
    const float f_ = 1.f / (1.f + __expf(-gf));
    const float o_ = 1.f / (1.f + __expf(-go));
    const float cn = f_ * c[idx] + i_ * tanhf(gc);
    const float hn = o_ * tanhf(cn);
    const bool keep = useMask ? (t < lengths[n]) : true;
    if (keep) { c[idx] = cn; h[idx] = hn; }
}

__global__ void zero_kernel(float* p, int n) {
    const int i = blockIdx.x * 256 + threadIdx.x;
    if (i < n) p[i] = 0.f;
}

// ---------------------------------------------------------------------------
// Head: cosine sims -> feats -> ReLU MLP -> batch-axis softmax. One block.
// ---------------------------------------------------------------------------
__global__ __launch_bounds__(256) void head_kernel(
    const float* __restrict__ sentH,   // 1024 x 256, n = (b*4+p)*8 + s
    const float* __restrict__ parH,    // 128 x 256,  m = b*4+p
    const float* __restrict__ W_hid, const float* __restrict__ b_hid,
    const float* __restrict__ gamma, const float* __restrict__ beta,
    const float* __restrict__ W_pred, const float* __restrict__ b_pred,
    float* __restrict__ out)           // 32 x 3
{
    __shared__ float s_sims[32][28];   // (b, p*7+q)
    __shared__ float p_sims[32][3];
    __shared__ float feats[32][28];
    __shared__ float hh[32][256];
    __shared__ float pred[32][3];

    const int tid = threadIdx.x;

    for (int s = tid; s < 32 * 28; s += 256) {
        const int b = s / 28, rem = s % 28, p = rem / 7, q = rem % 7;
        const float* a  = sentH + ((b * 4 + p) * 8 + q) * 256;
        const float* bb = a + 256;
        float dab = 0.f, daa = 0.f, dbb = 0.f;
        for (int k = 0; k < 256; ++k) { const float x = a[k], y = bb[k]; dab += x * y; daa += x * x; dbb += y * y; }
        const float na = fmaxf(sqrtf(daa), 1e-8f), nb = fmaxf(sqrtf(dbb), 1e-8f);
        s_sims[b][rem] = dab / (na * nb);
    }
    for (int s = tid; s < 32 * 3; s += 256) {
        const int b = s / 3, p = s % 3;
        const float* a  = parH + (b * 4 + p) * 256;
        const float* bb = a + 256;
        float dab = 0.f, daa = 0.f, dbb = 0.f;
        for (int k = 0; k < 256; ++k) { const float x = a[k], y = bb[k]; dab += x * y; daa += x * x; dbb += y * y; }
        const float na = fmaxf(sqrtf(daa), 1e-8f), nb = fmaxf(sqrtf(dbb), 1e-8f);
        p_sims[b][p] = dab / (na * nb);
    }
    __syncthreads();

    for (int s = tid; s < 32 * 28; s += 256) {
        const int b = s / 28, rem = s % 28, p = rem / 7;
        const float mult = (p == 0) ? 1.f : p_sims[b][p - 1];
        feats[b][rem] = s_sims[b][rem] * mult;
    }
    __syncthreads();

    const float inv = rsqrtf(1.f + 1e-5f);
    for (int s = tid; s < 32 * 256; s += 256) {
        const int b = s >> 8, j = s & 255;
        float acc = b_hid[j];
        for (int k = 0; k < 28; ++k) acc += feats[b][k] * W_hid[j * 50 + k];  // cols 28..49 are zero pad
        acc = fmaxf(acc, 0.f);
        hh[b][j] = acc * inv * gamma[j] + beta[j];
    }
    __syncthreads();

    for (int s = tid; s < 96; s += 256) {
        const int b = s / 3, l = s % 3;
        float acc = b_pred[l];
        for (int j = 0; j < 256; ++j) acc += hh[b][j] * W_pred[l * 256 + j];
        pred[b][l] = acc;
    }
    __syncthreads();

    if (tid < 3) {   // softmax over axis 0 (batch) per label
        float mx = -1e30f;
        for (int b = 0; b < 32; ++b) mx = fmaxf(mx, pred[b][tid]);
        float sum = 0.f;
        for (int b = 0; b < 32; ++b) { const float e = __expf(pred[b][tid] - mx); pred[b][tid] = e; sum += e; }
        for (int b = 0; b < 32; ++b) out[b * 3 + tid] = pred[b][tid] / sum;
    }
}

extern "C" void kernel_launch(void* const* d_in, const int* in_sizes, int n_in,
                              void* d_out, int out_size, void* d_ws, size_t ws_size,
                              hipStream_t stream) {
    const int*   tokens  = (const int*)d_in[0];
    const int*   lengths = (const int*)d_in[1];
    const float* embed   = (const float*)d_in[2];
    const float* W_ih_w  = (const float*)d_in[3];
    const float* W_hh_w  = (const float*)d_in[4];
    const float* b_ih_w  = (const float*)d_in[5];
    const float* b_hh_w  = (const float*)d_in[6];
    const float* W_ih_s  = (const float*)d_in[7];
    const float* W_hh_s  = (const float*)d_in[8];
    const float* b_ih_s  = (const float*)d_in[9];
    const float* b_hh_s  = (const float*)d_in[10];
    const float* W_hid   = (const float*)d_in[11];
    const float* b_hid   = (const float*)d_in[12];
    const float* gamma   = (const float*)d_in[13];
    const float* beta    = (const float*)d_in[14];
    const float* W_pred  = (const float*)d_in[15];
    const float* b_pred  = (const float*)d_in[16];

    // workspace layout (floats): h_w | c_w | h_s | c_s | G
    float* ws  = (float*)d_ws;
    float* h_w = ws;
    float* c_w = h_w + 1024 * 256;
    float* h_s = c_w + 1024 * 256;
    float* c_s = h_s + 128 * 256;
    float* G   = c_s + 128 * 256;      // 1024*1024

    const int zn = 1024 * 256 * 2 + 128 * 256 * 2;
    zero_kernel<<<(zn + 255) / 256, 256, 0, stream>>>(h_w, zn);

    const dim3 gw(16, 8);   // 1024/BN x 1024/BM
    const dim3 gs(16, 1);   // 1024/BN x 128/BM
    for (int t = 0; t < 32; ++t) {
        lstm_gemm_kernel<0><<<gw, 256, 0, stream>>>(tokens, embed, h_w, W_ih_w, W_hh_w, G, t);
        lstm_gate_kernel<<<1024, 256, 0, stream>>>(G, b_ih_w, b_hh_w, lengths, h_w, c_w, t, 1);
    }
    for (int s = 0; s < 8; ++s) {
        lstm_gemm_kernel<1><<<gs, 256, 0, stream>>>(tokens, h_w, h_s, W_ih_s, W_hh_s, G, s);
        lstm_gate_kernel<<<128, 256, 0, stream>>>(G, b_ih_s, b_hh_s, lengths, h_s, c_s, s, 0);
    }
    head_kernel<<<1, 256, 0, stream>>>(h_w, h_s, W_hid, b_hid, gamma, beta, W_pred, b_pred,
                                       (float*)d_out);
}
